// Attention_19808389169352
// MI455X (gfx1250) — matrix operand order repaired and checked by reading, not compiled, not run on hardware
//
#include <hip/hip_runtime.h>
#include <stdint.h>

// ---------------------------------------------------------------------------
// CDNA5 (gfx1250) attention forward: x@Wqkv+b -> flash-attn -> @Wproj+b
// Matmuls: V_WMMA_F32_16X16X32_BF16 (wave32).
// Global->LDS: GLOBAL_LOAD_ASYNC_TO_LDS_B128 (ASYNCcnt), double-buffered in
// both the GEMMs and the flash-attention inner loop. Softmax reductions via
// DPP8 / DPP16 row_half_mirror (pure VALU, no LDS round-trips).
// ---------------------------------------------------------------------------

typedef __attribute__((ext_vector_type(16))) __bf16 v16bf;
typedef __attribute__((ext_vector_type(8)))  float  v8f;

union Frag {
    v16bf    v;
    uint32_t u[8];
};

static __device__ inline v8f wmma_bf16(const Frag& a, const Frag& b, v8f c) {
    return __builtin_amdgcn_wmma_f32_16x16x32_bf16(false, a.v, false, b.v,
                                                   (short)0, c, false, false);
}

// --- cross-lane xor reductions over 16-lane halves (VALU-only) -------------
#define DPP8_XOR1 ((1) | (0 << 3) | (3 << 6) | (2 << 9) | (5 << 12) | (4 << 15) | (7 << 18) | (6 << 21))
#define DPP8_XOR2 ((2) | (3 << 3) | (0 << 6) | (1 << 9) | (6 << 12) | (7 << 15) | (4 << 18) | (5 << 21))
#define DPP8_XOR4 ((4) | (5 << 3) | (6 << 6) | (7 << 9) | (0 << 12) | (1 << 15) | (2 << 18) | (3 << 21))
#define DPP_ROW_HALF_MIRROR 0x141

static __device__ inline float red_max16(float x) {
    x = fmaxf(x, __int_as_float(__builtin_amdgcn_mov_dpp8(__float_as_int(x), DPP8_XOR1)));
    x = fmaxf(x, __int_as_float(__builtin_amdgcn_mov_dpp8(__float_as_int(x), DPP8_XOR2)));
    x = fmaxf(x, __int_as_float(__builtin_amdgcn_mov_dpp8(__float_as_int(x), DPP8_XOR4)));
    x = fmaxf(x, __int_as_float(__builtin_amdgcn_mov_dpp(__float_as_int(x),
                 DPP_ROW_HALF_MIRROR, 0xf, 0xf, true)));
    return x;
}
static __device__ inline float red_sum16(float x) {
    x += __int_as_float(__builtin_amdgcn_mov_dpp8(__float_as_int(x), DPP8_XOR1));
    x += __int_as_float(__builtin_amdgcn_mov_dpp8(__float_as_int(x), DPP8_XOR2));
    x += __int_as_float(__builtin_amdgcn_mov_dpp8(__float_as_int(x), DPP8_XOR4));
    x += __int_as_float(__builtin_amdgcn_mov_dpp(__float_as_int(x),
                 DPP_ROW_HALF_MIRROR, 0xf, 0xf, true));
    return x;
}

// --- async global->LDS copy helpers (CDNA5 path, ASYNCcnt-tracked) ---------
static __device__ inline void async_copy_32(uint32_t lds, uint64_t gaddr) {
    asm volatile("global_load_async_to_lds_b128 %0, %1, off\n\t"
                 "global_load_async_to_lds_b128 %0, %1, off offset:16"
                 :: "v"(lds), "v"(gaddr) : "memory");
}
static __device__ inline void async_copy_64(uint32_t lds, uint64_t gaddr) {
    asm volatile("global_load_async_to_lds_b128 %0, %1, off\n\t"
                 "global_load_async_to_lds_b128 %0, %1, off offset:16\n\t"
                 "global_load_async_to_lds_b128 %0, %1, off offset:32\n\t"
                 "global_load_async_to_lds_b128 %0, %1, off offset:48"
                 :: "v"(lds), "v"(gaddr) : "memory");
}
static __device__ inline void wait_async0() {
    asm volatile("s_wait_asynccnt 0x0" ::: "memory");
}
static __device__ inline uint32_t lds_off(const void* p) {
    return (uint32_t)(uintptr_t)p;   // generic LDS address: low 32 bits = LDS offset
}

// ---------------------------------------------------------------------------
// fp32 -> bf16 conversion
// ---------------------------------------------------------------------------
__global__ void f32_to_bf16_kernel(const float* __restrict__ in,
                                   __bf16* __restrict__ out, int n) {
    int i = blockIdx.x * blockDim.x + threadIdx.x;
    int stride = gridDim.x * blockDim.x;
    for (; i < n; i += stride) out[i] = (__bf16)in[i];
}

// ---------------------------------------------------------------------------
// GEMM: C[M,N] = A[M,K](bf16) @ B[K,N](bf16) + bias[N](f32)
// 256 thr / 8 waves; block tile 128x128; wave tile 32x64 (2x4 WMMA tiles).
// Double-buffered LDS: A tile via async-to-LDS, B tile transposed via regs.
// ---------------------------------------------------------------------------
template <typename OutT>
__global__ __launch_bounds__(256)
void gemm_bf16_wmma(const __bf16* __restrict__ A, const __bf16* __restrict__ B,
                    const float* __restrict__ bias, OutT* __restrict__ C,
                    int M, int N, int K) {
    __shared__ __bf16 As[2][128 * 32];   // [m][k]
    __shared__ __bf16 Bts[2][128 * 32];  // [n][k] (transposed)

    const int tid    = threadIdx.x;
    const int lane   = tid & 31;
    const int wave   = tid >> 5;
    const int wm     = wave & 3;
    const int wn     = wave >> 2;
    const int lane15 = lane & 15;
    const int laneHi = lane >> 4;
    const int koffA  = laneHi * 8;
    const int koffB  = laneHi * 8;

    const int blockM = blockIdx.x * 128;
    const int blockN = blockIdx.y * 128;

    const int arow  = tid >> 1;
    const int ahalf = (tid & 1) * 16;
    const int krow  = tid >> 3;
    const int nc    = (tid & 7) * 16;
    const __bf16* aSrc = A + (size_t)(blockM + arow) * K + ahalf;
    const __bf16* bSrc = B + (size_t)krow * N + blockN + nc;

    union BStage { uint4 q[2]; __bf16 h[16]; };
    BStage bst;

    v8f acc[2][4];
    #pragma unroll
    for (int nf = 0; nf < 4; ++nf) {
        float bv = bias[blockN + wn * 64 + nf * 16 + lane15];
        #pragma unroll
        for (int i = 0; i < 2; ++i)
            #pragma unroll
            for (int r = 0; r < 8; ++r) acc[i][nf][r] = bv;
    }

    // ---- prologue: fill buffer 0 ----
    async_copy_32(lds_off(&As[0][arow * 32 + ahalf]), (uint64_t)(uintptr_t)aSrc);
    bst.q[0] = *(const uint4*)(bSrc);
    bst.q[1] = *(const uint4*)(bSrc + 8);
    #pragma unroll
    for (int e = 0; e < 16; ++e)
        Bts[0][(nc + e) * 32 + krow] = bst.h[e];
    wait_async0();
    __syncthreads();

    int buf = 0;
    for (int k0 = 0; k0 < K; k0 += 32) {
        const int kn = k0 + 32;
        const bool hasNext = kn < K;

        // ---- prefetch next tile (overlaps with WMMA below) ----
        if (hasNext) {
            async_copy_32(lds_off(&As[buf ^ 1][arow * 32 + ahalf]),
                          (uint64_t)(uintptr_t)(aSrc + kn));
            const __bf16* bn = bSrc + (size_t)kn * N;
            bst.q[0] = *(const uint4*)(bn);
            bst.q[1] = *(const uint4*)(bn + 8);
        }

        // ---- compute on current buffer ----
        Frag a[2];
        #pragma unroll
        for (int i = 0; i < 2; ++i) {
            const __bf16* base = &As[buf][(wm * 32 + i * 16 + lane15) * 32];
            #pragma unroll
            for (int j = 0; j < 4; ++j) {
                a[i].u[j]     = *(const uint32_t*)(base + koffA + 2 * j);
                a[i].u[j + 4] = *(const uint32_t*)(base + 16 + koffA + 2 * j);
            }
        }
        #pragma unroll
        for (int nf = 0; nf < 4; ++nf) {
            Frag bf;
            const __bf16* base = &Bts[buf][(wn * 64 + nf * 16 + lane15) * 32 + koffB];
            #pragma unroll
            for (int j = 0; j < 8; ++j)
                bf.u[j] = *(const uint32_t*)(base + 2 * j + (j & 4) * 2);
            #pragma unroll
            for (int i = 0; i < 2; ++i)
                acc[i][nf] = wmma_bf16(a[i], bf, acc[i][nf]);
        }

        // ---- finish staging next buffer ----
        if (hasNext) {
            #pragma unroll
            for (int e = 0; e < 16; ++e)
                Bts[buf ^ 1][(nc + e) * 32 + krow] = bst.h[e];
            wait_async0();
        }
        __syncthreads();
        buf ^= 1;
    }

    // ---- epilogue: lane = column, 8 rows per lane ----
    #pragma unroll
    for (int i = 0; i < 2; ++i) {
        int rowBase = blockM + wm * 32 + i * 16 + laneHi * 8;
        #pragma unroll
        for (int nf = 0; nf < 4; ++nf) {
            int col = blockN + wn * 64 + nf * 16 + lane15;
            #pragma unroll
            for (int r = 0; r < 8; ++r)
                C[(size_t)(rowBase + r) * N + col] = (OutT)acc[i][nf][r];
        }
    }
}

// ---------------------------------------------------------------------------
// Flash attention (no scale, no mask). One block per (b, h, 64 q-rows).
// 128 thr = 4 waves; wave owns 16 q-rows. Online softmax fp32 with DPP
// reductions. Double-buffered K/V tiles: K via async-to-LDS, V transposed
// through registers; next-tile copy overlaps WMMA + softmax of current tile.
// ---------------------------------------------------------------------------
__global__ __launch_bounds__(128)
void flash_attn_bf16(const __bf16* __restrict__ qkv, __bf16* __restrict__ out,
                     int S, int D) {
    const int H = 16, HD = 64;
    __shared__ __bf16 Kt[2][64 * 64];     // [key][hd]
    __shared__ __bf16 Vt[2][64 * 64];     // [hd][key]
    __shared__ __bf16 Ps[4 * 16 * 64];    // per-wave P staging

    const int tid    = threadIdx.x;
    const int lane   = tid & 31;
    const int wave   = tid >> 5;
    const int lane15 = lane & 15;
    const int laneHi = lane >> 4;
    const int koffA  = laneHi * 8;
    const int koffB  = laneHi * 8;

    const int qt = blockIdx.x;
    const int bh = blockIdx.y;
    const int b  = bh / H;
    const int h  = bh % H;
    const size_t rowStride = (size_t)3 * D;
    const __bf16* qbase = qkv + (size_t)b * S * rowStride + h * HD;
    const __bf16* kbase = qbase + D;
    const __bf16* vbase = qbase + 2 * D;

    // Q fragments in registers (A layout, 2 K-steps over hd=64)
    Frag aq[2];
    {
        const __bf16* qrow =
            qbase + (size_t)(qt * 64 + wave * 16 + lane15) * rowStride;
        #pragma unroll
        for (int ks = 0; ks < 2; ++ks)
            #pragma unroll
            for (int j = 0; j < 4; ++j) {
                aq[ks].u[j]     = *(const uint32_t*)(qrow + ks * 32 + koffA + 2 * j);
                aq[ks].u[j + 4] = *(const uint32_t*)(qrow + ks * 32 + 16 + koffA + 2 * j);
            }
    }

    float m[8], l[8];
    v8f o[4];
    #pragma unroll
    for (int r = 0; r < 8; ++r) { m[r] = -1e30f; l[r] = 0.f; }
    #pragma unroll
    for (int nf = 0; nf < 4; ++nf)
        #pragma unroll
        for (int r = 0; r < 8; ++r) o[nf][r] = 0.f;

    __bf16* Pw = &Ps[wave * 16 * 64];
    const int key  = tid >> 1;
    const int half = (tid & 1) * 32;

    union VStage { uint4 q[4]; __bf16 hh[32]; };
    VStage vs;

    auto issueK = [&](int bufi, int kt) {
        async_copy_64(lds_off(&Kt[bufi][key * 64 + half]),
                      (uint64_t)(uintptr_t)(kbase + (size_t)(kt * 64 + key) * rowStride + half));
    };
    auto loadV = [&](int kt) {
        const __bf16* src = vbase + (size_t)(kt * 64 + key) * rowStride + half;
        #pragma unroll
        for (int q4 = 0; q4 < 4; ++q4)
            vs.q[q4] = *(const uint4*)(src + q4 * 8);
    };
    auto storeV = [&](int bufi) {
        #pragma unroll
        for (int e = 0; e < 32; ++e)
            Vt[bufi][(half + e) * 64 + key] = vs.hh[e];
    };

    // ---- prologue: fill buffer 0 ----
    issueK(0, 0);
    loadV(0);
    storeV(0);
    wait_async0();
    __syncthreads();

    const int NT = S / 64;
    int buf = 0;
    for (int kt = 0; kt < NT; ++kt) {
        const bool hasNext = (kt + 1) < NT;

        // ---- prefetch next K/V tile (overlaps with compute below) ----
        if (hasNext) {
            issueK(buf ^ 1, kt + 1);
            loadV(kt + 1);
        }

        // --- scores S = Q @ K^T ---
        v8f s[4];
        #pragma unroll
        for (int nf = 0; nf < 4; ++nf)
            #pragma unroll
            for (int r = 0; r < 8; ++r) s[nf][r] = 0.f;
        #pragma unroll
        for (int nf = 0; nf < 4; ++nf)
            #pragma unroll
            for (int ks = 0; ks < 2; ++ks) {
                Frag bk;
                const __bf16* base =
                    &Kt[buf][(nf * 16 + lane15) * 64 + ks * 32 + koffB];
                #pragma unroll
                for (int j = 0; j < 8; ++j)
                    bk.u[j] = *(const uint32_t*)(base + 2 * j + (j & 4) * 2);
                s[nf] = wmma_bf16(aq[ks], bk, s[nf]);
            }

        // --- online softmax (DPP reductions, pure VALU) ---
        #pragma unroll
        for (int r = 0; r < 8; ++r) {
            float mx = fmaxf(fmaxf(s[0][r], s[1][r]), fmaxf(s[2][r], s[3][r]));
            mx = red_max16(mx);
            float mn = fmaxf(m[r], mx);
            float sc = __expf(m[r] - mn);
            m[r] = mn;
            float rs = 0.f;
            #pragma unroll
            for (int nf = 0; nf < 4; ++nf) {
                float p = __expf(s[nf][r] - mn);
                s[nf][r] = p;
                rs += p;
            }
            rs = red_sum16(rs);
            l[r] = l[r] * sc + rs;
            #pragma unroll
            for (int nf = 0; nf < 4; ++nf) o[nf][r] *= sc;
        }

        // --- P (C-layout) -> LDS -> A-layout fragments ---
        #pragma unroll
        for (int nf = 0; nf < 4; ++nf)
            #pragma unroll
            for (int r = 0; r < 8; ++r)
                Pw[(laneHi * 8 + r) * 64 + nf * 16 + lane15] = (__bf16)s[nf][r];

        Frag pa[2];
        #pragma unroll
        for (int ks = 0; ks < 2; ++ks) {
            const __bf16* base = &Pw[lane15 * 64 + ks * 32];
            #pragma unroll
            for (int j = 0; j < 4; ++j) {
                pa[ks].u[j]     = *(const uint32_t*)(base + koffA + 2 * j);
                pa[ks].u[j + 4] = *(const uint32_t*)(base + 16 + koffA + 2 * j);
            }
        }

        // --- O += P @ V ---
        #pragma unroll
        for (int nf = 0; nf < 4; ++nf)
            #pragma unroll
            for (int ks = 0; ks < 2; ++ks) {
                Frag bv;
                const __bf16* base =
                    &Vt[buf][(nf * 16 + lane15) * 64 + ks * 32 + koffB];
                #pragma unroll
                for (int j = 0; j < 8; ++j)
                    bv.u[j] = *(const uint32_t*)(base + 2 * j + (j & 4) * 2);
                o[nf] = wmma_bf16(pa[ks], bv, o[nf]);
            }

        // ---- finish staging next buffer ----
        if (hasNext) {
            storeV(buf ^ 1);
            wait_async0();
        }
        __syncthreads();
        buf ^= 1;
    }

    // --- epilogue: normalize, merge heads, store bf16 ---
    #pragma unroll
    for (int r = 0; r < 8; ++r) {
        float rinv = 1.0f / l[r];
        int srow = qt * 64 + wave * 16 + laneHi * 8 + r;
        #pragma unroll
        for (int nf = 0; nf < 4; ++nf)
            out[((size_t)b * S + srow) * D + h * HD + nf * 16 + lane15] =
                (__bf16)(o[nf][r] * rinv);
    }
}

// ---------------------------------------------------------------------------
// Host-side orchestration
// ---------------------------------------------------------------------------
extern "C" void kernel_launch(void* const* d_in, const int* in_sizes, int n_in,
                              void* d_out, int out_size, void* d_ws, size_t ws_size,
                              hipStream_t stream) {
    const float* x      = (const float*)d_in[0];
    const float* w_attn = (const float*)d_in[1];
    const float* b_attn = (const float*)d_in[2];
    const float* w_proj = (const float*)d_in[3];
    const float* b_proj = (const float*)d_in[4];
    float* out = (float*)d_out;

    const int B = 2, S = 2048, D = 1024;
    const int M = B * S;           // 4096
    const int N3 = 3 * D;          // 3072

    char* ws = (char*)d_ws;
    __bf16* xb   = (__bf16*)ws; ws += (size_t)M * D * 2;
    __bf16* wab  = (__bf16*)ws; ws += (size_t)D * N3 * 2;
    __bf16* wpb  = (__bf16*)ws; ws += (size_t)D * D * 2;
    __bf16* qkvb = (__bf16*)ws; ws += (size_t)M * N3 * 2;
    __bf16* aob  = (__bf16*)ws;

    f32_to_bf16_kernel<<<2048, 256, 0, stream>>>(x, xb, M * D);
    f32_to_bf16_kernel<<<2048, 256, 0, stream>>>(w_attn, wab, D * N3);
    f32_to_bf16_kernel<<<1024, 256, 0, stream>>>(w_proj, wpb, D * D);

    dim3 gQKV(M / 128, N3 / 128);                   // 32 x 24
    gemm_bf16_wmma<__bf16><<<gQKV, 256, 0, stream>>>(xb, wab, b_attn, qkvb, M, N3, D);

    dim3 gATT(S / 64, B * 16);                      // 32 x 32
    flash_attn_bf16<<<gATT, 128, 0, stream>>>(qkvb, aob, S, D);

    dim3 gPRJ(M / 128, D / 128);                    // 32 x 8
    gemm_bf16_wmma<float><<<gPRJ, 256, 0, stream>>>(aob, wpb, b_proj, out, M, D, D);
}
